// MultiHeadAttentionOptimized_54795192763067
// MI455X (gfx1250) — compile-verified, matrix operand order checked by reading, not run on hardware
//
#include <hip/hip_runtime.h>

typedef __attribute__((ext_vector_type(16))) _Float16 v16h;
typedef __attribute__((ext_vector_type(8)))  float    v8f;

constexpr int BATCH = 2;
constexpr int SEQ   = 2048;
constexpr int EMB   = 1024;
constexpr int NHEAD = 16;
constexpr int HS    = 64;     // head size
constexpr int QKVC  = 3 * EMB;

// ISA 7.12.2: 16-bit A/B fragment K index for VGPR v, half h, lane-group grp.
__device__ __forceinline__ int kmap(int v, int grp) {
    return 2 * (v & 3) + ((v >> 2) << 4) + (grp << 3);
}

// Butterfly exchange within the 32-lane wave via ds_swizzle_b32 (group-of-32
// xor mode, immediate-encoded: offset = xor<<10 | and_mask 0x1F). Masks < 16
// keep the exchange inside each 16-lane half, which is what the 16-wide
// softmax reductions need.
template <int MASK>
__device__ __forceinline__ float swz_xor(float x) {
    return __int_as_float(
        __builtin_amdgcn_ds_swizzle(__float_as_int(x), (MASK << 10) | 0x1F));
}

// ---------------------------------------------------------------------------
// Tensor Data Mover (gfx1250). Builtin arity differs by toolchain:
//   ROCm 7.2 / clang-22 : 5 args (g0, g1, g2, g3, cpol)
//   amdgpu-toolchain / clang>=23 : 6 args (g0, g1, g2, g3, v8i extra, cpol)
// Round-2 histogram showed vmem unchanged with the 5-arg guard -> this driver
// is the clang>=23 lane, so select the 6-arg form there.
// ---------------------------------------------------------------------------
typedef __attribute__((ext_vector_type(4))) unsigned int u32x4;
typedef __attribute__((ext_vector_type(8))) int          i32x8;
typedef __attribute__((ext_vector_type(4))) int          i32x4;
typedef __attribute__((address_space(3))) const void*    lds_cptr_t;

__device__ __forceinline__ uint32_t lds_off32(const void* p) {
    return (uint32_t)(size_t)(lds_cptr_t)p;   // addrspacecast -> LDS byte offset
}

// 2D f16 tile DMA: tile_w elems per row, tile_h rows, row stride in elements.
// pad_interval_code / pad_amount_code per D# group1 encoding (ISA 8.4) chosen
// so the LDS row stride becomes tile_w + 8 halves (matches the LDS layouts).
__device__ __forceinline__ void tdm_load_2d_f16(
    const void* gptr, uint32_t lds_addr,
    uint32_t tile_w, uint32_t tile_h, uint32_t row_stride_elems,
    uint32_t pad_interval_code, uint32_t pad_amount_code)
{
    uint64_t ga = (uint64_t)(uintptr_t)gptr;
    u32x4 g0;
    g0[0] = 1u;                                     // count=1 (valid user D#)
    g0[1] = lds_addr;                               // LDS byte address
    g0[2] = (uint32_t)ga;                           // global_addr[31:0]
    g0[3] = ((uint32_t)(ga >> 32) & 0x1FFFFFFu) | (2u << 30);  // [56:32] | type=2
    i32x8 g1;
    g1[0] = (int)((1u << 16)                        // data_size = 2 bytes
                | (1u << 20)                        // pad_enable
                | (pad_interval_code << 22)
                | (pad_amount_code << 25));
    g1[1] = (int)(tile_w << 16);                    // tensor_dim0[15:0] @ 63:48
    g1[2] = (int)(tile_h << 16);                    // tensor_dim1[15:0] @ 95:80
    g1[3] = (int)(tile_w << 16);                    // tile_dim0       @ 127:112
    g1[4] = (int)tile_h;                            // tile_dim1       @ 143:128
    g1[5] = (int)row_stride_elems;                  // tensor_dim0_stride[31:0]
    g1[6] = 0;
    g1[7] = 0;
    i32x4 z4 = {0, 0, 0, 0};
#if defined(__clang_major__) && (__clang_major__ >= 23)
    i32x8 z8 = {0, 0, 0, 0, 0, 0, 0, 0};
    __builtin_amdgcn_tensor_load_to_lds(g0, g1, z4, z4, z8, 0);
#else
    __builtin_amdgcn_tensor_load_to_lds(g0, g1, z4, z4, 0);
#endif
}

// ---------------------------------------------------------------------------
// Tiled WMMA GEMM: C[M,N] = A[M,K] * B[K,N] (+bias). A f32 (convert) or f16
// (TDM DMA straight into LDS, overlapped with B staging). B f32, staged
// TRANSPOSED ([N][K], stride 40 halves) so B fragments are 16B runs ->
// ds_load_b128. Block: 256 threads = 8 waves; tile 128x64, BK=32.
// ---------------------------------------------------------------------------
template <typename AT, typename OT, bool BIAS>
__global__ __launch_bounds__(256) void gemm_f16wmma(
    const AT* __restrict__ A, const float* __restrict__ Bw,
    const float* __restrict__ bias, OT* __restrict__ Cout,
    int M, int N, int K)
{
    constexpr int BM = 128, BN = 64, BK = 32;
    constexpr int LDA = BK + 8, LDBT = BK + 8;      // 40 halves (16B-aligned runs)
    constexpr bool A_HALF = (sizeof(AT) == 2);
    __shared__ _Float16 As[BM * LDA];
    __shared__ _Float16 Bst[BN * LDBT];             // transposed B tile

    const int tid  = threadIdx.x;
    const int lane = tid & 31, warp = tid >> 5;
    const int grp  = lane >> 4, lcol = lane & 15;
    const int wm   = warp >> 1, wn = warp & 1;      // 4x2 warp grid
    const int bm   = blockIdx.y * BM, bn = blockIdx.x * BN;

    v8f acc[2][2] = {};

    for (int kk = 0; kk < K; kk += BK) {
        if (A_HALF && warp == 0) {   // one TDM issue per workgroup (EXEC ignored)
            tdm_load_2d_f16(A + (size_t)bm * K + kk, lds_off32(As),
                            BK, BM, (uint32_t)K, /*64B rows*/3, /*+16B*/3);
        }
        // Stage B tile transposed: Bst[n][k] (global reads coalesced along n).
        #pragma unroll
        for (int i = 0; i < (BK * BN) / 256; ++i) {
            int idx = tid + i * 256;
            int r = idx >> 6, c = idx & 63;          // r = k, c = n
            Bst[c * LDBT + r] = (_Float16)Bw[(size_t)(kk + r) * N + bn + c];
        }
        if (kk + BK < K) {  // gfx1250 global_prefetch_b8 for the next B tile
            __builtin_prefetch(&Bw[(size_t)(kk + BK) * N + bn + (tid & 63)], 0, 2);
        }
        if (!A_HALF) {
            // Cooperative A staging with f32 -> f16 conversion.
            #pragma unroll
            for (int i = 0; i < (BM * BK) / 256; ++i) {
                int idx = tid + i * 256;
                int r = idx >> 5, c = idx & 31;
                As[r * LDA + c] = (_Float16)A[(size_t)(bm + r) * K + kk + c];
            }
        } else if (warp == 0) {
            __builtin_amdgcn_s_wait_tensorcnt(0);    // DMA complete before barrier
        }
        __syncthreads();

        v16h af[2], bf[2];
        #pragma unroll
        for (int i = 0; i < 2; ++i) {
            int rbase = wm * 32 + i * 16 + lcol;
            #pragma unroll
            for (int v = 0; v < 8; ++v) {
                int k = kmap(v, grp);
                af[i][2 * v]     = As[rbase * LDA + k];
                af[i][2 * v + 1] = As[rbase * LDA + k + 1];
            }
        }
        #pragma unroll
        for (int j = 0; j < 2; ++j) {
            int cbase = wn * 32 + j * 16 + lcol;
            #pragma unroll
            for (int v = 0; v < 8; ++v) {
                int k = kmap(v, grp);
                bf[j][2 * v]     = Bst[cbase * LDBT + k];
                bf[j][2 * v + 1] = Bst[cbase * LDBT + k + 1];
            }
        }
        #pragma unroll
        for (int i = 0; i < 2; ++i)
            #pragma unroll
            for (int j = 0; j < 2; ++j)
                acc[i][j] = __builtin_amdgcn_wmma_f32_16x16x32_f16(
                    false, af[i], false, bf[j], (short)0, acc[i][j], false, false);
        __syncthreads();
    }

    #pragma unroll
    for (int i = 0; i < 2; ++i)
        #pragma unroll
        for (int j = 0; j < 2; ++j)
            #pragma unroll
            for (int v = 0; v < 8; ++v) {
                int gr = bm + wm * 32 + i * 16 + v + grp * 8;
                int gc = bn + wn * 32 + j * 16 + lcol;
                float val = acc[i][j][v];
                if (BIAS) val += bias[gc];
                Cout[(size_t)gr * N + gc] = (OT)val;
            }
}

// ---------------------------------------------------------------------------
// Flash attention (causal) over f16 QKV workspace.
// Block: 128 threads = 4 waves = 64 query rows of one (b,h).
// K tile (32x64 row-major) DMA'd by the TDM with padded rows; V tile staged
// TRANSPOSED ([dim][key], stride 40) so P*V B-fragments are ds_load_b128.
// ---------------------------------------------------------------------------
__global__ __launch_bounds__(128) void flash_attn_f16(
    const _Float16* __restrict__ qkv, _Float16* __restrict__ outh)
{
    constexpr int KB  = 32;          // keys per step
    constexpr int LKV = HS + 8;      // K tile row stride (matches TDM padding)
    constexpr int LVT = KB + 8;      // transposed V tile row stride
    __shared__ _Float16 Klds[KB * LKV];
    __shared__ _Float16 Vt[HS * LVT];
    __shared__ _Float16 Plds[4][16 * 40];   // per-wave P re-layout staging

    const int tid  = threadIdx.x;
    const int lane = tid & 31, wave = tid >> 5;
    const int grp  = lane >> 4, lcol = lane & 15;
    const int b    = blockIdx.z, h = blockIdx.y;
    const int q0b  = blockIdx.x * 64;
    const int q0   = q0b + wave * 16;

    // Q fragments (16x64 -> two 16x32 A-fragments), row = lcol.
    v16h qa[2];
    {
        const _Float16* qrow = qkv + (size_t)(b * SEQ + q0 + lcol) * QKVC + h * HS;
        #pragma unroll
        for (int f = 0; f < 2; ++f)
            #pragma unroll
            for (int v = 0; v < 8; ++v) {
                int k = f * 32 + kmap(v, grp);
                qa[f][2 * v]     = qrow[k];
                qa[f][2 * v + 1] = qrow[k + 1];
            }
    }

    float mrun[8], lrun[8];
    #pragma unroll
    for (int v = 0; v < 8; ++v) { mrun[v] = -3.0e38f; lrun[v] = 0.f; }
    v8f oacc[4] = {};

    const int kend = q0b + 64;       // causal horizon for the whole block
    for (int k0 = 0; k0 < kend; k0 += KB) {
        if (wave == 0) {             // K tile: 32 rows x 64 halves, +16B row pad
            tdm_load_2d_f16(qkv + (size_t)(b * SEQ + k0) * QKVC + EMB + h * HS,
                            lds_off32(Klds), HS, KB, QKVC, /*128B rows*/4, /*+16B*/3);
        }
        // V tile staged transposed: Vt[d][key] (global reads coalesced over d).
        #pragma unroll
        for (int i = 0; i < (KB * HS) / 128; ++i) {
            int idx = tid + i * 128;
            int kr = idx >> 6, d = idx & 63;
            Vt[d * LVT + kr] =
                qkv[(size_t)(b * SEQ + k0 + kr) * QKVC + 2 * EMB + h * HS + d];
        }
        if (wave == 0) __builtin_amdgcn_s_wait_tensorcnt(0);
        __syncthreads();

        if (k0 < q0 + 16) {   // wave-uniform causal skip -> EXEC stays all-1s
            // S = Q * K^T : two 16x16 n-tiles, K-dim 64 = 2 WMMA steps each.
            v8f s[2] = {};
            #pragma unroll
            for (int j = 0; j < 2; ++j) {
                v16h kb0, kb1;
                int key = j * 16 + lcol;
                #pragma unroll
                for (int v = 0; v < 8; ++v) {
                    int k = kmap(v, grp);
                    kb0[2 * v]     = Klds[key * LKV + k];
                    kb0[2 * v + 1] = Klds[key * LKV + k + 1];
                    kb1[2 * v]     = Klds[key * LKV + 32 + k];
                    kb1[2 * v + 1] = Klds[key * LKV + 32 + k + 1];
                }
                s[j] = __builtin_amdgcn_wmma_f32_16x16x32_f16(
                    false, qa[0], false, kb0, (short)0, s[j], false, false);
                s[j] = __builtin_amdgcn_wmma_f32_16x16x32_f16(
                    false, qa[1], false, kb1, (short)0, s[j], false, false);
            }

            // Scale, causal mask, online softmax (rows live at v + 8*grp).
            // Row-wide max/sum via immediate-encoded ds_swizzle butterflies.
            const float scale = 0.125f;   // 1/sqrt(64)
            float alpha[8];
            #pragma unroll
            for (int v = 0; v < 8; ++v) {
                int qr = q0 + v + grp * 8;
                #pragma unroll
                for (int j = 0; j < 2; ++j) {
                    int kc = k0 + j * 16 + lcol;
                    float sv = s[j][v] * scale;
                    s[j][v] = (kc > qr) ? -3.0e38f : sv;
                }
                float mx = fmaxf(s[0][v], s[1][v]);
                mx = fmaxf(mx, swz_xor<8>(mx));
                mx = fmaxf(mx, swz_xor<4>(mx));
                mx = fmaxf(mx, swz_xor<2>(mx));
                mx = fmaxf(mx, swz_xor<1>(mx));
                float mnew = fmaxf(mrun[v], mx);
                alpha[v] = __expf(mrun[v] - mnew);
                float p0 = __expf(s[0][v] - mnew);
                float p1 = __expf(s[1][v] - mnew);
                s[0][v] = p0; s[1][v] = p1;
                float sm = p0 + p1;
                sm += swz_xor<8>(sm);
                sm += swz_xor<4>(sm);
                sm += swz_xor<2>(sm);
                sm += swz_xor<1>(sm);
                lrun[v] = lrun[v] * alpha[v] + sm;
                mrun[v] = mnew;
            }

            // Re-layout P (C-layout -> A-fragment) through per-wave LDS patch.
            _Float16* pst = &Plds[wave][0];
            #pragma unroll
            for (int v = 0; v < 8; ++v) {
                int r = v + grp * 8;
                pst[r * 40 + lcol]      = (_Float16)s[0][v];
                pst[r * 40 + 16 + lcol] = (_Float16)s[1][v];
            }
            asm volatile("s_wait_dscnt 0" ::: "memory");  // wave-internal LDS RAW
            v16h pa;
            #pragma unroll
            for (int v = 0; v < 8; ++v) {
                int k = kmap(v, grp);
                pa[2 * v]     = pst[lcol * 40 + k];
                pa[2 * v + 1] = pst[lcol * 40 + k + 1];
            }

            // O = diag(alpha)*O + P * V : four 16x16 n-tiles, K-dim 32.
            #pragma unroll
            for (int j = 0; j < 4; ++j) {
                #pragma unroll
                for (int v = 0; v < 8; ++v) oacc[j][v] *= alpha[v];
                v16h vb;
                int cbase = j * 16 + lcol;
                #pragma unroll
                for (int v = 0; v < 8; ++v) {
                    int k = kmap(v, grp);
                    vb[2 * v]     = Vt[cbase * LVT + k];
                    vb[2 * v + 1] = Vt[cbase * LVT + k + 1];
                }
                oacc[j] = __builtin_amdgcn_wmma_f32_16x16x32_f16(
                    false, pa, false, vb, (short)0, oacc[j], false, false);
            }
        }
        __syncthreads();
    }

    // Normalize and store f16 to attention-out workspace [B*SEQ, EMB].
    #pragma unroll
    for (int v = 0; v < 8; ++v) {
        int r = v + grp * 8;
        float inv = 1.0f / lrun[v];
        size_t row = (size_t)(b * SEQ + q0 + r) * EMB + h * HS;
        #pragma unroll
        for (int j = 0; j < 4; ++j)
            outh[row + j * 16 + lcol] = (_Float16)(oacc[j][v] * inv);
    }
}

// ---------------------------------------------------------------------------
extern "C" void kernel_launch(void* const* d_in, const int* in_sizes, int n_in,
                              void* d_out, int out_size, void* d_ws, size_t ws_size,
                              hipStream_t stream)
{
    (void)in_sizes; (void)n_in; (void)out_size; (void)ws_size;
    const float* x    = (const float*)d_in[0];   // [2,2048,1024]
    const float* Wqkv = (const float*)d_in[1];   // [1024,3072]
    const float* Wout = (const float*)d_in[2];   // [1024,1024]
    const float* bout = (const float*)d_in[3];   // [1024]
    float* out = (float*)d_out;                  // [2,2048,1024] f32

    const int M = BATCH * SEQ;                   // 4096 rows
    _Float16* qkv_h  = (_Float16*)d_ws;                                   // 25.2 MB
    _Float16* attn_h = (_Float16*)((char*)d_ws +
                        (size_t)M * QKVC * sizeof(_Float16));             // +8.4 MB

    // 1) QKV projection: x[4096,1024] @ Wqkv[1024,3072] -> f16 qkv.
    gemm_f16wmma<float, _Float16, false>
        <<<dim3(QKVC / 64, M / 128), 256, 0, stream>>>(
            x, Wqkv, nullptr, qkv_h, M, QKVC, EMB);

    // 2) Fused causal flash attention over all (b, h, q-tile).
    flash_attn_f16<<<dim3(SEQ / 64, NHEAD, BATCH), 128, 0, stream>>>(qkv_h, attn_h);

    // 3) Output projection + bias: attn[4096,1024] @ Wout[1024,1024] -> f32.
    gemm_f16wmma<_Float16, float, true>
        <<<dim3(EMB / 64, M / 128), 256, 0, stream>>>(
            attn_h, Wout, bout, out, M, EMB, EMB);
}
